// SimpleGVPGNNModel_37220186587481
// MI455X (gfx1250) — matrix operand order repaired and checked by reading, not compiled, not run on hardware
//
#include <hip/hip_runtime.h>
#include <hip/hip_bf16.h>

// ---------------------------------------------------------------------------
// GVP-GNN scalar channel on CDNA5 (gfx1250), wave32, V_WMMA_F32_16X16X32_BF16.
// Vector channel (DV=3) is dead code w.r.t. the outputs and is skipped.
// Edge gather uses GLOBAL_LOAD_ASYNC_TO_LDS_B128 (ASYNCcnt) — CDNA5 async path.
// ---------------------------------------------------------------------------

typedef unsigned short u16;
typedef __bf16 v16bf __attribute__((ext_vector_type(16)));
typedef float  v8f   __attribute__((ext_vector_type(8)));
typedef unsigned int v4u __attribute__((ext_vector_type(4)));

#define MN   8192        // B*N nodes
#define EN   262144      // edges
#define LN   4           // layers
#define DS_  512
#define D2_  1024

__device__ __forceinline__ u16 f2bf(float f) {
  unsigned int u = __builtin_bit_cast(unsigned int, f);
  return (u16)((u + 0x7FFFu + ((u >> 16) & 1u)) >> 16);   // RNE
}

// Load 16 bf16 into a WMMA fragment from two 16-byte runs.
__device__ __forceinline__ v16bf ld_bf16_2x8(const u16* p0, const u16* p1) {
  union { v4u u[2]; v16bf v; } t;
  t.u[0] = *(const v4u*)p0;
  t.u[1] = *(const v4u*)p1;
  return t.v;
}

__device__ __forceinline__ v8f wmma_bf16(v16bf a, v16bf b, v8f c) {
  return __builtin_amdgcn_wmma_f32_16x16x32_bf16(
      /*neg_a=*/false, a, /*neg_b=*/false, b,
      /*c_mod=*/(short)0, c, /*reuse_a=*/false, /*reuse_b=*/false);
}

// Generic->LDS byte offset (LDS aperture keeps the offset in addr[31:0]).
__device__ __forceinline__ unsigned int lds_off(const void* p) {
  return (unsigned int)(unsigned long long)p;
}

// CDNA5 async memory->LDS copy of 16 bytes (no VGPR data path, ASYNCcnt).
__device__ __forceinline__ void async_copy_b128(unsigned int lds_byte_off,
                                                const void* gaddr) {
  asm volatile("global_load_async_to_lds_b128 %0, %1, off"
               :: "v"(lds_byte_off), "v"(gaddr)
               : "memory");
}

__device__ __forceinline__ void wait_asynccnt0() {
  asm volatile("s_wait_asynccnt 0x0" ::: "memory");
}

// ---------------------------------------------------------------------------
// Weight transpose + f32->bf16: src [K][N] (in,out) -> dst [N][K] so a WMMA
// B-fragment column is 64 contiguous bytes.
// ---------------------------------------------------------------------------
__global__ __launch_bounds__(256) void transpose_bf16_kernel(
    const float* __restrict__ src, u16* __restrict__ dst, int K, int N) {
  int idx = blockIdx.x * 256 + threadIdx.x;
  if (idx >= K * N) return;
  int n = idx / K;
  int k = idx - n * K;
  dst[idx] = f2bf(src[(size_t)k * N + n]);
}

// ---------------------------------------------------------------------------
// s = embed[token] * valid; keep f32 master + bf16 operand copy + mask.
// ---------------------------------------------------------------------------
__global__ __launch_bounds__(256) void init_kernel(
    const int* __restrict__ tok, const float* __restrict__ embed,
    float* __restrict__ s32, u16* __restrict__ s16, float* __restrict__ vm) {
  int q = blockIdx.x * 256 + threadIdx.x;     // MN*128 threads
  int node = q >> 7;
  int off  = (q & 127) << 2;
  int t = tok[node];
  float valid = (t != 0) ? 1.0f : 0.0f;
  float4 f = *(const float4*)(embed + (size_t)t * DS_ + off);
  f.x *= valid; f.y *= valid; f.z *= valid; f.w *= valid;
  *(float4*)(s32 + (size_t)node * DS_ + off) = f;
  uint2 p;
  p.x = (unsigned int)f2bf(f.x) | ((unsigned int)f2bf(f.y) << 16);
  p.y = (unsigned int)f2bf(f.z) | ((unsigned int)f2bf(f.w) << 16);
  *(uint2*)(s16 + (size_t)node * DS_ + off) = p;
  if (off == 0) vm[node] = valid;
}

// ---------------------------------------------------------------------------
// Edge kernel: 64 edges / workgroup, 8 waves.
//   A_lds[64][1024] = concat(s16[row], s16[col])   (async gather -> LDS)
//   H = silu(A @ W1t + b1)        (K=1024, N=1024)
//   es = (H @ W2t + b2) * emask   (K=1024, N=512)
//   atomicAdd into agg_s[row]
// ---------------------------------------------------------------------------
__global__ __launch_bounds__(256) void edge_gemm_kernel(
    const u16* __restrict__ sbf, const int* __restrict__ eidx, int E,
    const float* __restrict__ vmaskf,
    const u16* __restrict__ W1t, const float* __restrict__ b1,
    const u16* __restrict__ W2t, const float* __restrict__ b2,
    float* __restrict__ agg) {
  constexpr int AP = D2_ + 8;                  // padded LDS pitch (bank spread)
  extern __shared__ unsigned char smem[];
  u16*   A    = (u16*)smem;                    // [64][AP] bf16
  u16*   H    = A + 64 * AP;                   // [64][AP] bf16
  int*   ridx = (int*)(H + 64 * AP);
  int*   cidx = ridx + 64;
  float* em   = (float*)(cidx + 64);

  const int tid = threadIdx.x;
  const int e0  = blockIdx.x * 64;
  const v8f vzero = {0.f, 0.f, 0.f, 0.f, 0.f, 0.f, 0.f, 0.f};

  if (tid < 64) {
    int r = eidx[e0 + tid];
    int c = eidx[(size_t)E + e0 + tid];
    ridx[tid] = r;
    cidx[tid] = c;
    em[tid] = vmaskf[r] * vmaskf[c];
  }
  __syncthreads();

  // Async gather: 64*1024 bf16 = 8192 x 16B chunks, 32 per thread,
  // memory -> LDS directly (GLOBAL_LOAD_ASYNC_TO_LDS_B128).
  {
    const unsigned int a_base = lds_off(A);
    for (int i = 0; i < 32; ++i) {
      int q = tid + 256 * i;
      int m = q >> 7;                          // 128 chunks per row
      int off = (q & 127) << 3;
      const u16* src = (off < DS_)
          ? (sbf + (size_t)ridx[m] * DS_ + off)
          : (sbf + (size_t)cidx[m] * DS_ + (off - DS_));
      unsigned int dst = a_base + (unsigned int)((m * AP + off) * 2);
      async_copy_b128(dst, src);
    }
    wait_asynccnt0();
  }
  __syncthreads();

  const int w    = tid >> 5;
  const int lane = tid & 31;
  const int nsel = lane & 15;
  const int khA  = (lane & 16) >> 1;           // A K-half: 0 / 8
  const int khB  = (lane & 16) >> 1;           // B K-half: 0 / 8

  // ---- GEMM1: hidden = silu(A @ W1 + b1), each wave owns 8 n-tiles --------
  for (int j = 0; j < 8; ++j) {
    const int n = (w * 8 + j) * 16 + nsel;
    v8f acc[4];
#pragma unroll
    for (int mt = 0; mt < 4; ++mt) acc[mt] = vzero;
    const u16* bcol = W1t + (size_t)n * D2_ + khB;
    const u16* arow = A + nsel * AP + khA;
    for (int kb = 0; kb < 32; ++kb) {
      const u16* bp = bcol + kb * 32;
      v16bf bm = ld_bf16_2x8(bp, bp + 16);
#pragma unroll
      for (int mt = 0; mt < 4; ++mt) {
        const u16* ap = arow + mt * (16 * AP) + kb * 32;
        v16bf am = ld_bf16_2x8(ap, ap + 16);
        acc[mt] = wmma_bf16(am, bm, acc[mt]);
      }
    }
    const float bias = b1[n];
#pragma unroll
    for (int mt = 0; mt < 4; ++mt) {
#pragma unroll
      for (int g = 0; g < 8; ++g) {
        float x = acc[mt][g] + bias;
        x = x / (1.0f + __expf(-x));           // SiLU
        H[(mt * 16 + g + khA) * AP + n] = f2bf(x);
      }
    }
  }
  __syncthreads();

  // ---- GEMM2 + masked scatter-add, each wave owns 4 n-tiles ---------------
  for (int j = 0; j < 4; ++j) {
    const int n = (w * 4 + j) * 16 + nsel;
    v8f acc[4];
#pragma unroll
    for (int mt = 0; mt < 4; ++mt) acc[mt] = vzero;
    const u16* bcol = W2t + (size_t)n * D2_ + khB;
    const u16* arow = H + nsel * AP + khA;
    for (int kb = 0; kb < 32; ++kb) {
      const u16* bp = bcol + kb * 32;
      v16bf bm = ld_bf16_2x8(bp, bp + 16);
#pragma unroll
      for (int mt = 0; mt < 4; ++mt) {
        const u16* ap = arow + mt * (16 * AP) + kb * 32;
        v16bf am = ld_bf16_2x8(ap, ap + 16);
        acc[mt] = wmma_bf16(am, bm, acc[mt]);
      }
    }
    const float bias = b2[n];
#pragma unroll
    for (int mt = 0; mt < 4; ++mt) {
#pragma unroll
      for (int g = 0; g < 8; ++g) {
        int m = mt * 16 + g + khA;
        float val = (acc[mt][g] + bias) * em[m];
        atomicAdd(&agg[(size_t)ridx[m] * DS_ + n], val);
      }
    }
  }
}

// ---------------------------------------------------------------------------
// Update kernel: 64 nodes / workgroup.
//   s += silu(agg @ U1t + b1) @ U2t + b2   (512 -> 512 -> 512)
// ---------------------------------------------------------------------------
__global__ __launch_bounds__(256) void update_gemm_kernel(
    const float* __restrict__ agg,
    float* __restrict__ s32, u16* __restrict__ s16,
    const u16* __restrict__ U1t, const float* __restrict__ b1,
    const u16* __restrict__ U2t, const float* __restrict__ b2) {
  constexpr int AP = DS_ + 8;
  extern __shared__ unsigned char smem[];
  u16* A = (u16*)smem;                         // [64][AP] bf16
  u16* H = A + 64 * AP;                        // [64][AP] bf16

  const int tid   = threadIdx.x;
  const int node0 = blockIdx.x * 64;
  const v8f vzero = {0.f, 0.f, 0.f, 0.f, 0.f, 0.f, 0.f, 0.f};

  // Convert agg rows f32 -> bf16 into LDS: 64*512/4 = 8192 float4 chunks.
  for (int i = 0; i < 32; ++i) {
    int q = tid + 256 * i;
    int m = q >> 7;
    int off = (q & 127) << 2;
    float4 f = *(const float4*)(agg + (size_t)(node0 + m) * DS_ + off);
    uint2 p;
    p.x = (unsigned int)f2bf(f.x) | ((unsigned int)f2bf(f.y) << 16);
    p.y = (unsigned int)f2bf(f.z) | ((unsigned int)f2bf(f.w) << 16);
    *(uint2*)(A + m * AP + off) = p;
  }
  __syncthreads();

  const int w    = tid >> 5;
  const int lane = tid & 31;
  const int nsel = lane & 15;
  const int khA  = (lane & 16) >> 1;
  const int khB  = (lane & 16) >> 1;

  // ---- GEMM1 (K=512, N=512) ----------------------------------------------
  for (int j = 0; j < 4; ++j) {
    const int n = (w * 4 + j) * 16 + nsel;
    v8f acc[4];
#pragma unroll
    for (int mt = 0; mt < 4; ++mt) acc[mt] = vzero;
    const u16* bcol = U1t + (size_t)n * DS_ + khB;
    const u16* arow = A + nsel * AP + khA;
    for (int kb = 0; kb < 16; ++kb) {
      const u16* bp = bcol + kb * 32;
      v16bf bm = ld_bf16_2x8(bp, bp + 16);
#pragma unroll
      for (int mt = 0; mt < 4; ++mt) {
        const u16* ap = arow + mt * (16 * AP) + kb * 32;
        v16bf am = ld_bf16_2x8(ap, ap + 16);
        acc[mt] = wmma_bf16(am, bm, acc[mt]);
      }
    }
    const float bias = b1[n];
#pragma unroll
    for (int mt = 0; mt < 4; ++mt) {
#pragma unroll
      for (int g = 0; g < 8; ++g) {
        float x = acc[mt][g] + bias;
        x = x / (1.0f + __expf(-x));
        H[(mt * 16 + g + khA) * AP + n] = f2bf(x);
      }
    }
  }
  __syncthreads();

  // ---- GEMM2 (K=512, N=512) + residual into s ----------------------------
  for (int j = 0; j < 4; ++j) {
    const int n = (w * 4 + j) * 16 + nsel;
    v8f acc[4];
#pragma unroll
    for (int mt = 0; mt < 4; ++mt) acc[mt] = vzero;
    const u16* bcol = U2t + (size_t)n * DS_ + khB;
    const u16* arow = H + nsel * AP + khA;
    for (int kb = 0; kb < 16; ++kb) {
      const u16* bp = bcol + kb * 32;
      v16bf bm = ld_bf16_2x8(bp, bp + 16);
#pragma unroll
      for (int mt = 0; mt < 4; ++mt) {
        const u16* ap = arow + mt * (16 * AP) + kb * 32;
        v16bf am = ld_bf16_2x8(ap, ap + 16);
        acc[mt] = wmma_bf16(am, bm, acc[mt]);
      }
    }
    const float bias = b2[n];
#pragma unroll
    for (int mt = 0; mt < 4; ++mt) {
#pragma unroll
      for (int g = 0; g < 8; ++g) {
        int m = mt * 16 + g + khA;
        size_t idx = (size_t)(node0 + m) * DS_ + n;
        float sv = s32[idx] + acc[mt][g] + bias;
        s32[idx] = sv;
        s16[idx] = f2bf(sv);
      }
    }
  }
}

// ---------------------------------------------------------------------------
// Output: encoder_rep = s * vmask (f32), then padding mask appended.
// ---------------------------------------------------------------------------
__global__ __launch_bounds__(256) void final_kernel(
    const float* __restrict__ s32, const float* __restrict__ vm,
    const int* __restrict__ tok, float* __restrict__ out, int out_size) {
  int idx = blockIdx.x * 256 + threadIdx.x;
  const int body = MN * DS_;
  if (idx < body) {
    if (idx < out_size) out[idx] = s32[idx] * vm[idx >> 9];
  } else {
    int node = idx - body;
    if (node < MN && idx < out_size) out[idx] = (tok[node] == 0) ? 1.0f : 0.0f;
  }
}

// ---------------------------------------------------------------------------
extern "C" void kernel_launch(void* const* d_in, const int* in_sizes, int n_in,
                              void* d_out, int out_size, void* d_ws, size_t ws_size,
                              hipStream_t stream) {
  const int*   src_tokens = (const int*)d_in[0];
  const int*   edge_index = (const int*)d_in[2];
  const float* embed      = (const float*)d_in[3];
  const float* e_w1 = (const float*)d_in[6];
  const float* e_b1 = (const float*)d_in[7];
  const float* e_w2 = (const float*)d_in[8];
  const float* e_b2 = (const float*)d_in[9];
  const float* u_w1 = (const float*)d_in[14];
  const float* u_b1 = (const float*)d_in[15];
  const float* u_w2 = (const float*)d_in[16];
  const float* u_b2 = (const float*)d_in[17];
  const int E = in_sizes[2] / 2;               // = EN

  // ---- workspace carve ----------------------------------------------------
  char* p = (char*)d_ws;
  float* s32 = (float*)p;  p += (size_t)MN * DS_ * 4;
  u16*   s16 = (u16*)p;    p += (size_t)MN * DS_ * 2;
  float* vm  = (float*)p;  p += (size_t)MN * 4;
  float* agg = (float*)p;  p += (size_t)MN * DS_ * 4;
  u16*   W1t = (u16*)p;    p += (size_t)LN * D2_ * D2_ * 2;
  u16*   W2t = (u16*)p;    p += (size_t)LN * DS_ * D2_ * 2;
  u16*   U1t = (u16*)p;    p += (size_t)LN * DS_ * DS_ * 2;
  u16*   U2t = (u16*)p;    p += (size_t)LN * DS_ * DS_ * 2;

  const size_t SMEM_E = (size_t)2 * 64 * (D2_ + 8) * 2 + 64 * (4 + 4 + 4);
  const size_t SMEM_U = (size_t)2 * 64 * (DS_ + 8) * 2;
  (void)hipFuncSetAttribute((const void*)edge_gemm_kernel,
                            hipFuncAttributeMaxDynamicSharedMemorySize, (int)SMEM_E);
  (void)hipFuncSetAttribute((const void*)update_gemm_kernel,
                            hipFuncAttributeMaxDynamicSharedMemorySize, (int)SMEM_U);

  // ---- one-time weight transpose/convert (bf16, [out][in]) ----------------
  for (int l = 0; l < LN; ++l) {
    transpose_bf16_kernel<<<(D2_ * D2_) / 256, 256, 0, stream>>>(
        e_w1 + (size_t)l * D2_ * D2_, W1t + (size_t)l * D2_ * D2_, D2_, D2_);
    transpose_bf16_kernel<<<(D2_ * DS_) / 256, 256, 0, stream>>>(
        e_w2 + (size_t)l * D2_ * DS_, W2t + (size_t)l * DS_ * D2_, D2_, DS_);
    transpose_bf16_kernel<<<(DS_ * DS_) / 256, 256, 0, stream>>>(
        u_w1 + (size_t)l * DS_ * DS_, U1t + (size_t)l * DS_ * DS_, DS_, DS_);
    transpose_bf16_kernel<<<(DS_ * DS_) / 256, 256, 0, stream>>>(
        u_w2 + (size_t)l * DS_ * DS_, U2t + (size_t)l * DS_ * DS_, DS_, DS_);
  }

  // ---- init ---------------------------------------------------------------
  init_kernel<<<(MN * 128) / 256, 256, 0, stream>>>(src_tokens, embed, s32, s16, vm);

  // ---- layers -------------------------------------------------------------
  for (int l = 0; l < LN; ++l) {
    hipMemsetAsync(agg, 0, (size_t)MN * DS_ * 4, stream);
    edge_gemm_kernel<<<E / 64, 256, SMEM_E, stream>>>(
        s16, edge_index, E, vm,
        W1t + (size_t)l * D2_ * D2_, e_b1 + (size_t)l * D2_,
        W2t + (size_t)l * DS_ * D2_, e_b2 + (size_t)l * DS_,
        agg);
    update_gemm_kernel<<<MN / 64, 256, SMEM_U, stream>>>(
        agg, s32, s16,
        U1t + (size_t)l * DS_ * DS_, u_b1 + (size_t)l * DS_,
        U2t + (size_t)l * DS_ * DS_, u_b2 + (size_t)l * DS_);
  }

  // ---- output -------------------------------------------------------------
  const int total = MN * DS_ + MN;
  final_kernel<<<(total + 255) / 256, 256, 0, stream>>>(s32, vm, src_tokens,
                                                        (float*)d_out, out_size);
}
